// GENELink_43224550868106
// MI455X (gfx1250) — compile-verified, matrix operand order checked by reading, not run on hardware
//
#include <hip/hip_runtime.h>
#include <hip/hip_bf16.h>

typedef __attribute__((ext_vector_type(16))) __bf16 v16bf;
typedef __attribute__((ext_vector_type(8)))  float  v8f;

union FragU { uint4 q[2]; v16bf v; };

__device__ inline unsigned short f32_to_bf16u(float f) {
    unsigned int u = __float_as_uint(f);
    u += 0x7FFFu + ((u >> 16) & 1u);           // round-to-nearest-even
    return (unsigned short)(u >> 16);
}

__device__ inline v8f wmma_bf16(v16bf a, v16bf b, v8f c) {
    return __builtin_amdgcn_wmma_f32_16x16x32_bf16(false, a, false, b, (short)0, c, false, false);
}

// 32-bit LDS byte offset for async-to-LDS dst operand
__device__ inline unsigned lds_offset(const void* p) {
    return (unsigned)(unsigned long long)(__attribute__((address_space(3))) const char*)p;
}

// ---------------------------------------------------------------- converts
__global__ void cvt_bf16(const float* __restrict__ in, unsigned short* __restrict__ out, int n) {
    int i = blockIdx.x * blockDim.x + threadIdx.x;
    if (i < n) out[i] = f32_to_bf16u(in[i]);
}
// in: [K,N] f32  ->  out: [N,K] bf16   (weight transpose so B-fragments are contiguous)
__global__ void cvtT_bf16(const float* __restrict__ in, unsigned short* __restrict__ out, int K, int N) {
    int n = blockIdx.x * 16 + threadIdx.x;
    int k = blockIdx.y * 16 + threadIdx.y;
    if (n < N && k < K) out[(size_t)n * K + k] = f32_to_bf16u(in[(size_t)k * N + n]);
}

// ---------------------------------------------------------------- GEMM (bf16 WMMA, f32 acc, LDS-free)
// C[M,N] = act(A[M,K] @ Bt[N,K]^T + bias).  Block = 16 rows x 128 cols, 8 waves,
// one 16x16 WMMA tile per wave; every fragment = two contiguous global_load_b128.
__global__ __launch_bounds__(256) void gemm_bf16(
    const unsigned short* __restrict__ A,   // [M,K] bf16 row-major
    const unsigned short* __restrict__ Bt,  // [N,K] bf16 (B transposed)
    float* __restrict__ Cf,                 // [M,N] f32   (nullable)
    unsigned short* __restrict__ Cb,        // [M,N] bf16  (nullable)
    unsigned short* __restrict__ Cbt,       // [N,M] bf16 transposed (nullable)
    const float* __restrict__ bias, float slope, int act,
    int M, int N, int K)
{
    const int tid  = threadIdx.x;
    const int lane = tid & 31, wave = tid >> 5;
    const int row0 = blockIdx.x * 16;
    const int n0   = blockIdx.y * 128;
    const int kb4  = (lane >> 4) * 8;       // per-lane K sub-base (ISA 16-bit A/B layout)
    const int rA   = lane & 15;
    const int colL = n0 + wave * 16 + (lane & 15);
    const unsigned short* pa = A  + (size_t)(row0 + rA) * K + kb4;
    const unsigned short* pb = Bt + (size_t)colL * K + kb4;
    v8f acc = {};

    #pragma unroll 4
    for (int kb = 0; kb < K; kb += 32) {
        FragU fa, fb;
        fa.q[0] = *(const uint4*)(pa + kb);
        fa.q[1] = *(const uint4*)(pa + kb + 16);
        fb.q[0] = *(const uint4*)(pb + kb);
        fb.q[1] = *(const uint4*)(pb + kb + 16);
        acc = wmma_bf16(fa.v, fb.v, acc);
    }

    const int rbase = (lane >> 4) * 8;      // C layout: VGPR v -> row v (+8 for hi lanes)
    const float bcol = bias ? bias[colL] : 0.0f;
    float val[8];
    #pragma unroll
    for (int v = 0; v < 8; v++) {
        float x = acc[v] + bcol;
        if (act) x = x > 0.0f ? x : x * slope;
        val[v] = x;
    }
    if (Cf) {
        #pragma unroll
        for (int v = 0; v < 8; v++)
            Cf[(size_t)(row0 + rbase + v) * N + colL] = val[v];
    }
    if (Cb) {
        #pragma unroll
        for (int v = 0; v < 8; v++)
            Cb[(size_t)(row0 + rbase + v) * N + colL] = f32_to_bf16u(val[v]);
    }
    if (Cbt) { // lane holds 8 consecutive rows of one column -> one packed 16B store
        union { uint4 q; unsigned short s[8]; } pk;
        #pragma unroll
        for (int v = 0; v < 8; v++) pk.s[v] = f32_to_bf16u(val[v]);
        *(uint4*)(Cbt + (size_t)colL * M + row0 + rbase) = pk.q;
    }
}

// ---------------------------------------------------------------- q/k projections (rank-1 logits)
__global__ __launch_bounds__(256) void qk_kernel(
    const float* __restrict__ V, const float* __restrict__ avec,
    float* __restrict__ q, float* __restrict__ k, int N, int D)
{
    int tid = threadIdx.x, lane = tid & 31, wave = tid >> 5;
    int row = blockIdx.x * 8 + wave;
    if (row >= N) return;
    const float* vr = V + (size_t)row * D;
    float qs = 0.0f, ks = 0.0f;
    for (int d = lane; d < D; d += 32) {
        float v = vr[d];
        qs += v * avec[d];
        ks += v * avec[D + d];
    }
    #pragma unroll
    for (int m = 16; m >= 1; m >>= 1) {
        qs += __shfl_xor(qs, m, 32);
        ks += __shfl_xor(ks, m, 32);
    }
    if (lane == 0) { q[row] = qs; k[row] = ks; }
}

// ---------------------------------------------------------------- fused flash GAT attention
// One block = 16 rows x full 256-col head output.  Streams 32-col tiles:
// adj tile double-buffered via async global->LDS; masked lrelu logits ->
// online softmax -> bf16 P tile written in A-fragment order -> P@V via WMMA
// with B fragments as contiguous 16B loads from V^T.  Epilogue fuses lrelu,
// row L2-normalize, +bias, optional ELU.
__global__ __launch_bounds__(256) void gat_attn(
    const float* __restrict__ qv, const float* __restrict__ kv,
    const unsigned short* __restrict__ Vt,  // [256][N] bf16 (V transposed)
    const int* __restrict__ adj,
    const float* __restrict__ bias,
    float* __restrict__ outF, unsigned short* __restrict__ outB,
    int N, int headOff, int outStride, int applyElu)
{
    __shared__ float sQ[16], sM[16], sL[16], sScale[16], sSq[16];
    __shared__ float sE[16 * 32];
    __shared__ __align__(32) unsigned short sP[32 * 16];   // A-fragment order: [lane][t]
    __shared__ __align__(16) int sAdj[2][16 * 32];

    const int tid  = threadIdx.x;
    const int lane = tid & 31, wave = tid >> 5;
    const int row0 = blockIdx.x * 16;
    const int kb4  = (lane >> 4) * 8;
    const int rbase = (lane >> 4) * 8;
    const int c0   = wave * 32 + (lane & 15);
    const unsigned short* pb0 = Vt + (size_t)c0 * N + kb4;
    const unsigned short* pb1 = Vt + (size_t)(c0 + 16) * N + kb4;

    if (tid < 16) {
        sQ[tid] = qv[row0 + tid];
        sM[tid] = -9.0e15f; sL[tid] = 0.0f; sSq[tid] = 0.0f;
    }
    v8f acc0 = {}, acc1 = {};

    // async prefetch of adj tile jb=0 into buffer 0 (threads 0..127, 16B each)
    if (tid < 128) {
        const int r = tid >> 3, c4 = (tid & 7) * 4;
        const int* g = adj + (size_t)(row0 + r) * N + c4;
        unsigned loff = lds_offset(&sAdj[0][r * 32 + c4]);
        asm volatile("global_load_async_to_lds_b128 %0, %1, off" :: "v"(loff), "v"(g) : "memory");
    }
    __syncthreads();

    for (int jb = 0; jb < N; jb += 32) {
        const int buf = (jb >> 5) & 1;
        asm volatile("s_wait_asynccnt 0x0" ::: "memory");
        __syncthreads();                      // adj tile `buf` visible to all waves
        if (jb + 32 < N && tid < 128) {       // overlap: fetch next tile into other buffer
            const int r = tid >> 3, c4 = (tid & 7) * 4;
            const int* g = adj + (size_t)(row0 + r) * N + jb + 32 + c4;
            unsigned loff = lds_offset(&sAdj[buf ^ 1][r * 32 + c4]);
            asm volatile("global_load_async_to_lds_b128 %0, %1, off" :: "v"(loff), "v"(g) : "memory");
        }
        // masked leaky-relu logits, 2 per thread
        #pragma unroll
        for (int e = 0; e < 2; e++) {
            int l = tid + e * 256;
            int r = l >> 5, c = l & 31;
            float ev = sQ[r] + kv[jb + c];
            ev = ev > 0.0f ? ev : 0.2f * ev;
            if (sAdj[buf][l] == 0) ev = -9.0e15f;
            sE[l] = ev;
        }
        __syncthreads();

        if (tid < 16) { // online softmax update, one row per thread; P in fragment order
            float mold = sM[tid], mmax = mold;
            #pragma unroll
            for (int c = 0; c < 32; c++) mmax = fmaxf(mmax, sE[tid * 32 + c]);
            float scale = __expf(mold - mmax);
            float lsum = 0.0f;
            #pragma unroll
            for (int c = 0; c < 32; c++) {
                float p = __expf(sE[tid * 32 + c] - mmax);
                // ISA A 16x32 layout: lane = row + 16*((c>>3)&1), t = (c&7) + 8*(c>>4)
                sP[(tid + (((c >> 3) & 1) << 4)) * 16 + (c & 7) + ((c >> 4) << 3)] = f32_to_bf16u(p);
                lsum += p;
            }
            sL[tid] = sL[tid] * scale + lsum;
            sM[tid] = mmax;
            sScale[tid] = scale;
        }
        __syncthreads();

        #pragma unroll
        for (int v = 0; v < 8; v++) { // rescale running accumulators
            float sc = sScale[rbase + v];
            acc0[v] *= sc; acc1[v] *= sc;
        }
        v16bf af = *(const v16bf*)&sP[lane * 16];   // 2x ds_load_b128, contiguous
        FragU f0, f1;                               // V^T fragments: contiguous 16B global loads
        f0.q[0] = *(const uint4*)(pb0 + jb);
        f0.q[1] = *(const uint4*)(pb0 + jb + 16);
        f1.q[0] = *(const uint4*)(pb1 + jb);
        f1.q[1] = *(const uint4*)(pb1 + jb + 16);
        acc0 = wmma_bf16(af, f0.v, acc0);
        acc1 = wmma_bf16(af, f1.v, acc1);
        __syncthreads();
    }

    // epilogue: softmax-normalize, lrelu(0.2), row L2 norm, +bias, optional ELU
    float o0[8], o1[8];
    #pragma unroll
    for (int v = 0; v < 8; v++) {
        float inv = 1.0f / sL[rbase + v];
        float a = acc0[v] * inv, b = acc1[v] * inv;
        a = a > 0.0f ? a : 0.2f * a;
        b = b > 0.0f ? b : 0.2f * b;
        o0[v] = a; o1[v] = b;
    }
    #pragma unroll
    for (int v = 0; v < 8; v++)
        atomicAdd(&sSq[rbase + v], o0[v] * o0[v] + o1[v] * o1[v]);  // ds_add_f32
    __syncthreads();

    const int col = wave * 32 + (lane & 15);
    const float bc0 = bias[col], bc1 = bias[col + 16];
    #pragma unroll
    for (int v = 0; v < 8; v++) {
        int gr = row0 + rbase + v;
        float inv = 1.0f / fmaxf(sqrtf(sSq[rbase + v]), 1e-12f);
        float a = o0[v] * inv + bc0;
        float b = o1[v] * inv + bc1;
        if (applyElu) {
            a = a > 0.0f ? a : __expf(a) - 1.0f;
            b = b > 0.0f ? b : __expf(b) - 1.0f;
        }
        size_t i0 = (size_t)gr * outStride + headOff + col;
        if (outF) { outF[i0] = a; outF[i0 + 16] = b; }
        if (outB) { outB[i0] = f32_to_bf16u(a); outB[i0 + 16] = f32_to_bf16u(b); }
    }
}

// ---------------------------------------------------------------- pair gather + 256->2 linear
__global__ void pair_kernel(const float* __restrict__ tf, const float* __restrict__ tg,
                            const long long* __restrict__ samp,
                            const float* __restrict__ linW, const float* __restrict__ linb,
                            float* __restrict__ out, int B)
{
    int s = blockIdx.x * blockDim.x + threadIdx.x;
    if (s >= B) return;
    long long i = samp[2 * s], j = samp[2 * s + 1];
    const float* fr = tf + (size_t)i * 128;
    const float* gr = tg + (size_t)j * 128;
    float a0 = linb[0], a1 = linb[1];
    #pragma unroll 4
    for (int d = 0; d < 128; d++) {
        float fv = fr[d];
        a0 += fv * linW[d * 2];     a1 += fv * linW[d * 2 + 1];
    }
    #pragma unroll 4
    for (int d = 0; d < 128; d++) {
        float gv = gr[d];
        a0 += gv * linW[(128 + d) * 2]; a1 += gv * linW[(128 + d) * 2 + 1];
    }
    out[2 * s] = a0; out[2 * s + 1] = a1;
}

// ---------------------------------------------------------------- driver
extern "C" void kernel_launch(void* const* d_in, const int* in_sizes, int n_in,
                              void* d_out, int out_size, void* d_ws, size_t ws_size,
                              hipStream_t stream)
{
    const int N = 6144, DIN = 512, D = 256, HEADS = 2, HID = 512, H3 = 512, DOUT = 128, NB = 16384;
    const float* x     = (const float*)d_in[0];
    const int*   adj   = (const int*)d_in[1];
    const long long* samp = (const long long*)d_in[2];
    const float* W1   = (const float*)d_in[3];
    const float* a1   = (const float*)d_in[4];
    const float* b1   = (const float*)d_in[5];
    const float* W2   = (const float*)d_in[6];
    const float* a2   = (const float*)d_in[7];
    const float* b2   = (const float*)d_in[8];
    const float* tf1W = (const float*)d_in[9];
    const float* tf1b = (const float*)d_in[10];
    const float* tf2W = (const float*)d_in[11];
    const float* tf2b = (const float*)d_in[12];
    const float* tg1W = (const float*)d_in[13];
    const float* tg1b = (const float*)d_in[14];
    const float* tg2W = (const float*)d_in[15];
    const float* tg2b = (const float*)d_in[16];
    const float* linW = (const float*)d_in[17];
    const float* linb = (const float*)d_in[18];
    float* pred = (float*)d_out;

    char* w = (char*)d_ws;
    size_t off = 0;
    auto alloc = [&](size_t bytes) -> void* {
        void* p = w + off; off += (bytes + 255) & ~(size_t)255; return p;
    };
    unsigned short* xb    = (unsigned short*)alloc((size_t)N * DIN * 2);   // x bf16 [N,DIN]
    unsigned short* w1t   = (unsigned short*)alloc((size_t)HEADS * D * DIN * 2); // per-head [D,DIN]
    unsigned short* w2t   = (unsigned short*)alloc((size_t)HEADS * D * HID * 2); // per-head [D,HID]
    unsigned short* tf1t  = (unsigned short*)alloc((size_t)H3 * HID * 2);
    unsigned short* tf2t  = (unsigned short*)alloc((size_t)DOUT * H3 * 2);
    unsigned short* tg1t  = (unsigned short*)alloc((size_t)H3 * HID * 2);
    unsigned short* tg2t  = (unsigned short*)alloc((size_t)DOUT * H3 * 2);
    float*          Vf    = (float*)alloc((size_t)N * D * 4);              // V f32 [N,D]
    unsigned short* Vt    = (unsigned short*)alloc((size_t)D * N * 2);     // V^T bf16 [D,N]
    float*          qv    = (float*)alloc((size_t)N * 4);
    float*          kvv   = (float*)alloc((size_t)N * 4);
    unsigned short* hB    = (unsigned short*)alloc((size_t)N * HID * 2);
    unsigned short* eB    = (unsigned short*)alloc((size_t)N * HID * 2);
    unsigned short* t1B   = (unsigned short*)alloc((size_t)N * H3 * 2);
    float*          tfF   = (float*)alloc((size_t)N * DOUT * 4);
    float*          tgF   = (float*)alloc((size_t)N * DOUT * 4);

    cvt_bf16<<<(unsigned)(((size_t)N * DIN + 255) / 256), 256, 0, stream>>>(x, xb, N * DIN);
    dim3 tblk(16, 16);
    for (int h = 0; h < HEADS; h++) {
        cvtT_bf16<<<dim3(D / 16, DIN / 16), tblk, 0, stream>>>(W1 + (size_t)h * DIN * D,
                                                               w1t + (size_t)h * D * DIN, DIN, D);
        cvtT_bf16<<<dim3(D / 16, HID / 16), tblk, 0, stream>>>(W2 + (size_t)h * HID * D,
                                                               w2t + (size_t)h * D * HID, HID, D);
    }
    cvtT_bf16<<<dim3(H3 / 16, HID / 16), tblk, 0, stream>>>(tf1W, tf1t, HID, H3);
    cvtT_bf16<<<dim3(DOUT / 16, H3 / 16), tblk, 0, stream>>>(tf2W, tf2t, H3, DOUT);
    cvtT_bf16<<<dim3(H3 / 16, HID / 16), tblk, 0, stream>>>(tg1W, tg1t, HID, H3);
    cvtT_bf16<<<dim3(DOUT / 16, H3 / 16), tblk, 0, stream>>>(tg2W, tg2t, H3, DOUT);

    dim3 blk(256);
    // GAT layer 1 (ELU fused into attention epilogue)
    for (int h = 0; h < HEADS; h++) {
        gemm_bf16<<<dim3(N / 16, D / 128), blk, 0, stream>>>(
            xb, w1t + (size_t)h * D * DIN, Vf, nullptr, Vt, nullptr, 0.0f, 0, N, D, DIN);
        qk_kernel<<<N / 8, blk, 0, stream>>>(Vf, a1 + (size_t)h * 2 * D, qv, kvv, N, D);
        gat_attn<<<N / 16, blk, 0, stream>>>(qv, kvv, Vt, adj, b1 + (size_t)h * D,
                                             nullptr, hB, N, h * D, HID, 1);
    }
    // GAT layer 2
    for (int h = 0; h < HEADS; h++) {
        gemm_bf16<<<dim3(N / 16, D / 128), blk, 0, stream>>>(
            hB, w2t + (size_t)h * D * HID, Vf, nullptr, Vt, nullptr, 0.0f, 0, N, D, HID);
        qk_kernel<<<N / 8, blk, 0, stream>>>(Vf, a2 + (size_t)h * 2 * D, qv, kvv, N, D);
        gat_attn<<<N / 16, blk, 0, stream>>>(qv, kvv, Vt, adj, b2 + (size_t)h * D,
                                             nullptr, eB, N, h * D, HID, 0);
    }
    // MLP heads (bias + leaky-relu fused into GEMM epilogues)
    gemm_bf16<<<dim3(N / 16, H3 / 128),   blk, 0, stream>>>(eB,  tf1t, nullptr, t1B, nullptr, tf1b, 0.01f, 1, N, H3,   HID);
    gemm_bf16<<<dim3(N / 16, DOUT / 128), blk, 0, stream>>>(t1B, tf2t, tfF, nullptr, nullptr, tf2b, 0.01f, 1, N, DOUT, H3);
    gemm_bf16<<<dim3(N / 16, H3 / 128),   blk, 0, stream>>>(eB,  tg1t, nullptr, t1B, nullptr, tg1b, 0.01f, 1, N, H3,   HID);
    gemm_bf16<<<dim3(N / 16, DOUT / 128), blk, 0, stream>>>(t1B, tg2t, tgF, nullptr, nullptr, tg2b, 0.01f, 1, N, DOUT, H3);
    // pair gather + final linear
    pair_kernel<<<(NB + 255) / 256, blk, 0, stream>>>(tfF, tgF, samp, linW, linb, pred, NB);

    (void)in_sizes; (void)n_in; (void)out_size; (void)ws_size;
}
